// MultiHead_Attention_60284160967280
// MI455X (gfx1250) — matrix operand order repaired and checked by reading, not compiled, not run on hardware
//
#include <hip/hip_runtime.h>

typedef __attribute__((ext_vector_type(16))) _Float16 v16h;
typedef __attribute__((ext_vector_type(8)))  _Float16 v8h;
typedef __attribute__((ext_vector_type(8)))  float    v8f;

constexpr int Bb   = 4;
constexpr int Ss   = 2048;
constexpr int Din  = 768;
constexpr int Dout = 768;
constexpr int Hh   = 12;
constexpr int Dh   = 64;
constexpr int Mrows = Bb * Ss;        // 8192

template <bool V> struct BoolC { static constexpr bool value = V; };

__device__ __forceinline__ v8f wmma_f16(v16h a, v16h b, v8f c) {
  // D = A(16x32 f16) * B(32x16 f16) + C(16x16 f32)
  return __builtin_amdgcn_wmma_f32_16x16x32_f16(
      /*neg_a=*/false, a, /*neg_b=*/false, b,
      /*c_mod=*/(short)0, c, /*reuse_a=*/false, /*reuse_b=*/false);
}

// ---- DPP16 row-rotate helpers: log-reduction within each 16-lane row, pure VALU
template <int CTRL>
__device__ __forceinline__ float dpp16_f(float x) {
  return __int_as_float(__builtin_amdgcn_update_dpp(
      __float_as_int(x), __float_as_int(x), CTRL, 0xF, 0xF, true));
}
__device__ __forceinline__ float row16_max(float x) {
  x = fmaxf(x, dpp16_f<0x121>(x));   // row_ror:1
  x = fmaxf(x, dpp16_f<0x122>(x));   // row_ror:2
  x = fmaxf(x, dpp16_f<0x124>(x));   // row_ror:4
  x = fmaxf(x, dpp16_f<0x128>(x));   // row_ror:8
  return x;
}
__device__ __forceinline__ float row16_sum(float x) {
  x += dpp16_f<0x121>(x);
  x += dpp16_f<0x122>(x);
  x += dpp16_f<0x124>(x);
  x += dpp16_f<0x128>(x);
  return x;
}

// ---------------------------------------------------------------------------
// x f32 -> f16, vectorized by 8
// ---------------------------------------------------------------------------
__global__ __launch_bounds__(256)
void xconv_kernel(const float* __restrict__ X, _Float16* __restrict__ Xh, int n) {
  const int idx = (blockIdx.x * blockDim.x + threadIdx.x) * 8;
  if (idx >= n) return;
  v8f v = *(const v8f*)(X + idx);
  v8h o;
#pragma unroll
  for (int i = 0; i < 8; ++i) o[i] = (_Float16)v[i];
  *(v8h*)(Xh + idx) = o;
}

// ---------------------------------------------------------------------------
// Weight transpose+convert: Wt[n][k] = (f16) W[k][n]   (K x N -> N x K)
// ---------------------------------------------------------------------------
__global__ __launch_bounds__(256)
void wtrans_kernel(const float* __restrict__ W, _Float16* __restrict__ Wt,
                   int K, int N) {
  int idx = blockIdx.x * blockDim.x + threadIdx.x;
  if (idx >= K * N) return;
  int n = idx / K;
  int k = idx - n * K;
  Wt[(size_t)n * K + k] = (_Float16)W[(size_t)k * N + n];
}

// ---------------------------------------------------------------------------
// WMMA GEMM: one wave computes a 32x64 tile of A[M,K](f16) @ Bt[N,K]^T(f16).
// Every B fragment is reused for 2 WMMAs (2 M-subtiles). Inner loop is pure
// load -> wmma (no conversions).
// MODE 0: store f16 * scale as [B,H,S,DH]   (Q with scale=1/sqrt(DH), K with 1)
// MODE 1: store f16 as [B,H,DH,S]           (V, transposed)
// MODE 2: store f32 [M,N] with bias         (output projection)
// ---------------------------------------------------------------------------
template <int MODE>
__global__ __launch_bounds__(256)
void gemm32x64_kernel(const _Float16* __restrict__ A,
                      const _Float16* __restrict__ Bt,
                      const float* __restrict__ bias,
                      void* __restrict__ Out,
                      int M, int K, int N, float scale) {
  const int lane = threadIdx.x & 31;
  const int wid  = blockIdx.x * (blockDim.x >> 5) + (threadIdx.x >> 5);
  const int tilesN = N >> 6;
  const int tm = wid / tilesN;
  const int tn = wid - tm * tilesN;
  if (tm * 32 >= M) return;
  const int ln = lane & 15;
  const int kh = lane >> 4;          // which 16-lane half

  v8f acc[2][4] = {};

  for (int k0 = 0; k0 < K; k0 += 32) {
    // ---- A fragments (2 x 16x32)
    v16h a[2];
#pragma unroll
    for (int mt = 0; mt < 2; ++mt) {
      const _Float16* Ah = A + (size_t)(tm * 32 + mt * 16 + ln) * K;
      v8h lo = *(const v8h*)(Ah + k0 + kh * 16);
      v8h hi = *(const v8h*)(Ah + k0 + kh * 16 + 8);
#pragma unroll
      for (int i = 0; i < 8; ++i) { a[mt][i] = lo[i]; a[mt][8 + i] = hi[i]; }
    }
    // ---- 4 B fragments, each reused across both M-subtiles
#pragma unroll
    for (int j = 0; j < 4; ++j) {
      const int ncol = tn * 64 + j * 16 + ln;
      v16h b = *(const v16h*)(Bt + (size_t)ncol * K + k0 + kh * 16);
#pragma unroll
      for (int mt = 0; mt < 2; ++mt)
        acc[mt][j] = wmma_f16(a[mt], b, acc[mt][j]);
    }
  }

  // ---- store. C layout: row = tm*32 + mt*16 + kh*8 + r, col = tn*64 + j*16 + ln
#pragma unroll
  for (int mt = 0; mt < 2; ++mt) {
#pragma unroll
    for (int j = 0; j < 4; ++j) {
      const int ncol = tn * 64 + j * 16 + ln;
#pragma unroll
      for (int r = 0; r < 8; ++r) {
        const int rowg = tm * 32 + mt * 16 + kh * 8 + r;
        const float val = acc[mt][j][r] * scale;
        if (MODE == 2) {
          ((float*)Out)[(size_t)rowg * N + ncol] = val + bias[ncol];
        } else {
          const int bidx = rowg >> 11;           // / S
          const int s    = rowg & (Ss - 1);
          const int h    = ncol >> 6;            // / DH
          const int dh   = ncol & (Dh - 1);
          if (MODE == 0) {
            ((_Float16*)Out)[(((size_t)bidx * Hh + h) * Ss + s) * Dh + dh] = (_Float16)val;
          } else {
            ((_Float16*)Out)[(((size_t)bidx * Hh + h) * Dh + dh) * Ss + s] = (_Float16)val;
          }
        }
      }
    }
  }
}

// ---------------------------------------------------------------------------
// Flash attention: one wave per (b,h, 32-row q tile). kv tiles of 32.
// Q (pre-scaled by 1/sqrt(DH)), K: f16 [B,H,S,DH]; Vt: f16 [B,H,DH,S];
// ctx out: f16 [B,S,H*DH]. Causal masking only executed on the diagonal tile.
// ---------------------------------------------------------------------------
__global__ __launch_bounds__(256)
void flash_attn_kernel(const _Float16* __restrict__ Q,
                       const _Float16* __restrict__ Km,
                       const _Float16* __restrict__ Vt,
                       _Float16* __restrict__ ctx) {
  __shared__ _Float16 plds[8][32 * 32];     // per-wave P staging (2 KB each)
  const int lane = threadIdx.x & 31;
  const int w    = threadIdx.x >> 5;
  const int gw   = blockIdx.x * 8 + w;      // 0 .. B*H*(S/32)-1
  const int qt   = gw & (Ss / 32 - 1);      // q tile index (S/32 = 64, pow2)
  const int bh   = gw >> 6;                 // b*H + h
  const int ln = lane & 15;
  const int kh = lane >> 4;

  const _Float16* Qh = Q  + (size_t)bh * Ss * Dh;
  const _Float16* Kh = Km + (size_t)bh * Ss * Dh;
  const _Float16* Vh = Vt + (size_t)bh * Dh * Ss;

  // ---- Q A-fragments: [m-subtile][dh step]
  v16h qa[2][2];
#pragma unroll
  for (int mt = 0; mt < 2; ++mt) {
#pragma unroll
    for (int s2 = 0; s2 < 2; ++s2) {
      const _Float16* qr = Qh + (size_t)(qt * 32 + mt * 16 + ln) * Dh + s2 * 32;
      v8h lo = *(const v8h*)(qr + kh * 16);
      v8h hi = *(const v8h*)(qr + kh * 16 + 8);
#pragma unroll
      for (int i = 0; i < 8; ++i) { qa[mt][s2][i] = lo[i]; qa[mt][s2][8 + i] = hi[i]; }
    }
  }

  float m_i[2][8], l_i[2][8];
#pragma unroll
  for (int mt = 0; mt < 2; ++mt)
#pragma unroll
    for (int r = 0; r < 8; ++r) { m_i[mt][r] = -1e30f; l_i[mt][r] = 0.f; }
  v8f acc[2][4] = {};

  const int ntiles = qt + 1;                // causal: kv tiles up to the diagonal
  for (int t = 0; t < ntiles; ++t) {
    const int kv0 = t * 32;
    if (t + 1 < ntiles) {   // prefetch next K/V tiles (global_prefetch_b8)
      __builtin_prefetch(Kh + (size_t)(kv0 + 32 + lane) * Dh, 0, 0);
      __builtin_prefetch(Vh + (size_t)lane * Ss + kv0 + 32, 0, 0);
    }
    // ---- scores S = Q @ K^T  (32 x 32, four 16x16 fragments)
    v8f sf[2][2] = {};                      // [m-subtile][n-subtile]
#pragma unroll
    for (int nt = 0; nt < 2; ++nt) {
      const _Float16* kr = Kh + (size_t)(kv0 + nt * 16 + ln) * Dh;  // lane = kv row
#pragma unroll
      for (int s2 = 0; s2 < 2; ++s2) {
        v16h kb = *(const v16h*)(kr + s2 * 32 + kh * 16);
#pragma unroll
        for (int mt = 0; mt < 2; ++mt)
          sf[mt][nt] = wmma_f16(qa[mt][s2], kb, sf[mt][nt]);
      }
    }
    // ---- online softmax (row = qt*32 + mt*16 + kh*8 + r, col = lane&15).
    // Masking only needed on the diagonal tile; selected by a uniform branch.
    auto softmax_step = [&](auto MASKC) {
      constexpr bool MASK = decltype(MASKC)::value;
#pragma unroll
      for (int mt = 0; mt < 2; ++mt) {
#pragma unroll
        for (int r = 0; r < 8; ++r) {
          float s0 = sf[mt][0][r];
          float s1 = sf[mt][1][r];
          if (MASK) {
            const int rloc = mt * 16 + kh * 8 + r;   // row within 32-tile
            s0 = (ln      > rloc) ? -1e30f : s0;     // diag tile: col > row
            s1 = (16 + ln > rloc) ? -1e30f : s1;
          }
          const float mx    = row16_max(fmaxf(s0, s1));
          const float mnew  = fmaxf(m_i[mt][r], mx);
          const float alpha = __expf(m_i[mt][r] - mnew);
          const float p0 = __expf(s0 - mnew);
          const float p1 = __expf(s1 - mnew);
          const float ps = row16_sum(p0 + p1);
          l_i[mt][r] = l_i[mt][r] * alpha + ps;
          m_i[mt][r] = mnew;
          sf[mt][0][r] = p0;
          sf[mt][1][r] = p1;
#pragma unroll
          for (int j = 0; j < 4; ++j) acc[mt][j][r] *= alpha;
        }
      }
    };
    if (t == qt) softmax_step(BoolC<true>{});
    else         softmax_step(BoolC<false>{});

    // ---- stage P (32x32 f16) in LDS to relayout C-frag -> A-frag
    _Float16* pl = &plds[w][0];
#pragma unroll
    for (int mt = 0; mt < 2; ++mt)
#pragma unroll
      for (int nt = 0; nt < 2; ++nt)
#pragma unroll
        for (int r = 0; r < 8; ++r)
          pl[(mt * 16 + kh * 8 + r) * 32 + nt * 16 + ln] = (_Float16)sf[mt][nt][r];
    asm volatile("s_wait_dscnt 0" ::: "memory");
    v16h pa[2];
#pragma unroll
    for (int mt = 0; mt < 2; ++mt) {
      v8h lo = *(const v8h*)(pl + (mt * 16 + ln) * 32 + kh * 16);
      v8h hi = *(const v8h*)(pl + (mt * 16 + ln) * 32 + kh * 16 + 8);
#pragma unroll
      for (int i = 0; i < 8; ++i) { pa[mt][i] = lo[i]; pa[mt][8 + i] = hi[i]; }
    }
    // ---- acc += P(32x32) @ V(32x64); each V fragment reused for 2 WMMAs
#pragma unroll
    for (int j = 0; j < 4; ++j) {
      const _Float16* vr = Vh + (size_t)(j * 16 + ln) * Ss + kv0 + kh * 16;
      v16h vb = *(const v16h*)vr;
#pragma unroll
      for (int mt = 0; mt < 2; ++mt)
        acc[mt][j] = wmma_f16(pa[mt], vb, acc[mt][j]);
    }
  }

  // ---- epilogue: normalize, write ctx as f16 [B,S,H*DH]
  const int bidx = bh / Hh;
  const int h    = bh - bidx * Hh;
#pragma unroll
  for (int mt = 0; mt < 2; ++mt) {
    const float inv0 = 1.f;   // placeholder to keep loop simple
    (void)inv0;
#pragma unroll
    for (int j = 0; j < 4; ++j) {
#pragma unroll
      for (int r = 0; r < 8; ++r) {
        const int s = qt * 32 + mt * 16 + kh * 8 + r;
        const float o = acc[mt][j][r] / l_i[mt][r];
        ctx[((size_t)bidx * Ss + s) * Dout + h * Dh + j * 16 + ln] = (_Float16)o;
      }
    }
  }
}

// ---------------------------------------------------------------------------
extern "C" void kernel_launch(void* const* d_in, const int* in_sizes, int n_in,
                              void* d_out, int out_size, void* d_ws, size_t ws_size,
                              hipStream_t stream) {
  (void)in_sizes; (void)n_in; (void)out_size; (void)ws_size;
  const float* x  = (const float*)d_in[0];
  // d_in[1] = causal mask: not needed (computed analytically)
  const float* Wq = (const float*)d_in[2];
  const float* Wk = (const float*)d_in[3];
  const float* Wv = (const float*)d_in[4];
  const float* Wo = (const float*)d_in[5];
  const float* bo = (const float*)d_in[6];
  float* out = (float*)d_out;

  char* ws = (char*)d_ws;
  size_t off = 0;
  auto alloc = [&](size_t bytes) -> void* {
    void* p = ws + off;
    off = (off + bytes + 255) & ~(size_t)255;
    return p;
  };
  _Float16* Wqt = (_Float16*)alloc((size_t)Din * Dout * 2);
  _Float16* Wkt = (_Float16*)alloc((size_t)Din * Dout * 2);
  _Float16* Wvt = (_Float16*)alloc((size_t)Din * Dout * 2);
  _Float16* Wot = (_Float16*)alloc((size_t)Dout * Dout * 2);
  _Float16* Xh  = (_Float16*)alloc((size_t)Mrows * Din * 2);
  _Float16* Qw  = (_Float16*)alloc((size_t)Bb * Hh * Ss * Dh * 2);
  _Float16* Kw  = (_Float16*)alloc((size_t)Bb * Hh * Ss * Dh * 2);
  _Float16* Vw  = (_Float16*)alloc((size_t)Bb * Hh * Ss * Dh * 2);
  _Float16* Cw  = (_Float16*)alloc((size_t)Mrows * Dout * 2);

  const dim3 blk(256);
  const int xc_blocks = (Mrows * Din / 8 + 255) / 256;      // 3072
  xconv_kernel<<<xc_blocks, blk, 0, stream>>>(x, Xh, Mrows * Din);

  const int wt_blocks = (Din * Dout + 255) / 256;
  wtrans_kernel<<<wt_blocks, blk, 0, stream>>>(Wq, Wqt, Din, Dout);
  wtrans_kernel<<<wt_blocks, blk, 0, stream>>>(Wk, Wkt, Din, Dout);
  wtrans_kernel<<<wt_blocks, blk, 0, stream>>>(Wv, Wvt, Din, Dout);
  wtrans_kernel<<<wt_blocks, blk, 0, stream>>>(Wo, Wot, Dout, Dout);

  const float qscale = 0.125f;   // 1/sqrt(DH): folded into Q
  const int gemm_blocks = (Mrows / 32) * (Dout / 64) / 8;   // 384
  gemm32x64_kernel<0><<<gemm_blocks, blk, 0, stream>>>(Xh, Wqt, nullptr, Qw, Mrows, Din, Dout, qscale);
  gemm32x64_kernel<0><<<gemm_blocks, blk, 0, stream>>>(Xh, Wkt, nullptr, Kw, Mrows, Din, Dout, 1.0f);
  gemm32x64_kernel<1><<<gemm_blocks, blk, 0, stream>>>(Xh, Wvt, nullptr, Vw, Mrows, Din, Dout, 1.0f);

  const int fa_blocks = (Bb * Hh * (Ss / 32)) / 8;          // 384
  flash_attn_kernel<<<fa_blocks, blk, 0, stream>>>(Qw, Kw, Vw, Cw);

  gemm32x64_kernel<2><<<gemm_blocks, blk, 0, stream>>>(Cw, Wot, bo, out, Mrows, Dout, Dout, 1.0f);
}
